// Linear2Bit_34342558498841
// MI455X (gfx1250) — matrix operand order repaired and checked by reading, not compiled, not run on hardware
//
#include <hip/hip_runtime.h>

typedef __attribute__((ext_vector_type(16))) _Float16 v16h;
typedef __attribute__((ext_vector_type(8)))  _Float16 v8h;
typedef __attribute__((ext_vector_type(4)))  _Float16 v4h;
typedef __attribute__((ext_vector_type(2)))  _Float16 v2h;
typedef __attribute__((ext_vector_type(8)))  float    v8f;
typedef __attribute__((ext_vector_type(4)))  float    v4f;
typedef __attribute__((ext_vector_type(4)))  int      v4i;

namespace {
constexpr int IN_F   = 4096;
constexpr int OUT_F  = 8192;
constexpr int M_TOT  = 256;
constexpr int BM     = 64;            // block tile M
constexpr int BN     = 128;           // block tile N
constexpr int BK     = 64;            // K step per iteration (2 wmma k-steps)
constexpr int KTILES = IN_F / BK;     // 64
constexpr int ASTR   = BK + 8;        // 72 halfs (144B) padded stride -> bank-conflict free
constexpr int BSTR   = BK + 8;
constexpr int KGROUPS = IN_F / 16;    // 256 quant groups per output row
}

union Frag  { v16h v; v8h h[2]; v2h p[8]; };
union Pack2 { unsigned u; v2h h; };
union Pack4 { v4h v4; v2h v2[2]; };

__device__ __forceinline__ v2h bits_to_h2(unsigned bits) {
    Pack2 t; t.u = bits; return t.h;
}

__device__ __forceinline__ v2h cvt_pk_f16(float lo, float hi) {
    return __builtin_bit_cast(v2h, __builtin_amdgcn_cvt_pkrtz(lo, hi));
}

__global__ __launch_bounds__(256)
void linear2bit_wmma_kernel(const float* __restrict__ x,
                            const int* __restrict__ wq,
                            const _Float16* __restrict__ wnorm,
                            const float* __restrict__ bias,
                            float* __restrict__ out)
{
    __shared__ _Float16 As[2][BM * ASTR];   // x tile, f16
    __shared__ _Float16 Bs[2][BN * BSTR];   // dequantized W tile, f16 ([n][k])

    const int t    = threadIdx.x;
    const int lane = t & 31;
    const int wave = t >> 5;                // 0..7
    const int m0   = blockIdx.y * BM;
    const int n0   = blockIdx.x * BN;
    const int tm   = wave & 3;              // wave's M sub-tile (0..3)
    const int nh   = wave >> 2;             // wave's N half (0..1): n-tiles nh*4..nh*4+3

    v8f      acc[4] = {};
    v4f      aReg[4];
    v4i      bReg[2];
    _Float16 nReg[2];

    // packed-f16 decode constant
    const v2h neg1024 = bits_to_h2(0xE400E400u);   // {-1024, -1024}

    // running 32-bit per-thread offsets -> saddr+voffset global loads,
    // one v_add_nc_u32 per stream per iteration instead of 64-bit recompute
    int aOff[4], wOff[2], nOff[2];
#pragma unroll
    for (int s = 0; s < 4; ++s) {
        int q   = t + 256 * s;              // 0..1023 float4 chunks
        int row = q >> 4;                   // 64 rows
        int c4  = q & 15;                   // 16 float4 per row
        aOff[s] = (m0 + row) * IN_F + c4 * 4;
    }
#pragma unroll
    for (int s = 0; s < 2; ++s) {
        int g  = t + 256 * s;               // 0..511 groups per tile
        int nl = g >> 2;                    // 128 rows (N)
        int kg = g & 3;                     // 4 k-groups of 16
        int gg = (n0 + nl) * KGROUPS + kg;
        wOff[s] = gg * 4;                   // int32 index into wq
        nOff[s] = gg;                       // index into wnorm
    }

    // ---- global -> registers; advances offsets to the next tile ----
    auto load_tile = [&]() {
#pragma unroll
        for (int s = 0; s < 4; ++s) {
            aReg[s] = *(const v4f*)(x + aOff[s]);
            aOff[s] += BK;
        }
#pragma unroll
        for (int s = 0; s < 2; ++s) {
            bReg[s] = *(const v4i*)(wq + wOff[s]);
            wOff[s] += 16;                  // 4 groups * 4 int32
            nReg[s] = wnorm[nOff[s]];
            nOff[s] += 4;
        }
        // offsets now point at the next tile: speculative prefetch
        // (OOB at the tail is silently dropped per prefetch semantics)
        __builtin_prefetch(x + aOff[0], 0, 0);
        __builtin_prefetch(wq + wOff[0], 0, 0);
    };

    // ---- registers -> LDS (convert / dequantize, packed f16 path) ----
    auto stage_tile = [&](int buf) {
#pragma unroll
        for (int s = 0; s < 4; ++s) {
            int q   = t + 256 * s;
            int row = q >> 4;
            int c4  = q & 15;
            Pack4 h;                               // v_cvt_pk_rtz_f16_f32 x2
            h.v2[0] = cvt_pk_f16(aReg[s].x, aReg[s].y);
            h.v2[1] = cvt_pk_f16(aReg[s].z, aReg[s].w);
            *(v4h*)&As[buf][row * ASTR + c4 * 4] = h.v4;
        }
#pragma unroll
        for (int s = 0; s < 2; ++s) {
            int g  = t + 256 * s;
            int nl = g >> 2;
            int kg = g & 3;
            const _Float16 nh16 = nReg[s];
            const _Float16 sh   = nh16 * (_Float16)(2.0f / 3.0f);
            const v2h scale2 = {sh, sh};           // { 2n/3, 2n/3 }
            const v2h negn2  = {-nh16, -nh16};     // { -n, -n }
            Frag d;
#pragma unroll
            for (int j = 0; j < 4; ++j) {
                // p holds one byte: elements 4j+i = (p >> 2i) & 3
                unsigned p = (unsigned)bReg[s][j];
                // pack 2-bit fields into halves of a dword, bias into f16 exponent:
                // 0x6400|u == f16(1024+u) exactly
                unsigned b01 = ((p & 0x3u)        | ((p & 0xCu)  << 14)) | 0x64006400u;
                unsigned b23 = (((p >> 4) & 0x3u) | ((p & 0xC0u) << 10)) | 0x64006400u;
                // (1024+u) - 1024 is exact in f16 -> u ; then u*(2n/3) - n
                v2h u01 = bits_to_h2(b01) + neg1024;       // v_pk_add_f16
                v2h u23 = bits_to_h2(b23) + neg1024;
                d.p[2 * j]     = u01 * scale2 + negn2;     // v_pk_fma_f16
                d.p[2 * j + 1] = u23 * scale2 + negn2;
            }
            _Float16* dst = &Bs[buf][nl * BSTR + kg * 16];
            *(v8h*)dst       = d.h[0];
            *(v8h*)(dst + 8) = d.h[1];
        }
    };

    // ---- LDS -> fragments -> v_wmma_f32_16x16x32_f16 ----
    auto compute_tile = [&](int buf) {
#pragma unroll
        for (int kk = 0; kk < BK; kk += 32) {
            // A 16x32 f16 layout: lane L -> M=L%16; VGPR0-3: K=(L/16)*8+0..7,
            //                                      VGPR4-7: K=16+(L/16)*8+0..7
            Frag a;
            const _Float16* ab =
                &As[buf][(tm * 16 + (lane & 15)) * ASTR + kk + (lane >> 4) * 8];
            a.h[0] = *(const v8h*)ab;
            a.h[1] = *(const v8h*)(ab + 16);
            // pre-stage all 4 B fragments so the ds_loads issue as one burst and
            // the WMMAs retire against partial s_wait_dscnt counts
            Frag b[4];
#pragma unroll
            for (int tn = 0; tn < 4; ++tn) {
                // B 32x16 f16 layout: lane L -> N=L%16, K=(L/16)*8+0..7, then 16+(L/16)*8+0..7
                const _Float16* bb =
                    &Bs[buf][((nh * 4 + tn) * 16 + (lane & 15)) * BSTR + kk + (lane >> 4) * 8];
                b[tn].h[0] = *(const v8h*)bb;
                b[tn].h[1] = *(const v8h*)(bb + 16);
            }
#pragma unroll
            for (int tn = 0; tn < 4; ++tn) {
                acc[tn] = __builtin_amdgcn_wmma_f32_16x16x32_f16(
                    false, a.v, false, b[tn].v, (short)0, acc[tn], false, false);
            }
        }
    };

    // ---- pipelined main loop ----
    load_tile();            // tile 0
    stage_tile(0);
    __syncthreads();

#pragma unroll 2
    for (int kt = 0; kt < KTILES; ++kt) {
        const int cur = kt & 1;
        if (kt + 1 < KTILES) load_tile();         // global loads in flight over WMMA
        compute_tile(cur);
        if (kt + 1 < KTILES) stage_tile(cur ^ 1); // dequant + LDS store into other buffer
        __syncthreads();
    }

    // ---- epilogue: C layout lanes 0-15 -> M=e, lanes 16-31 -> M=8+e; N = lane%16 ----
    const int row0 = m0 + tm * 16 + (lane >> 4) * 8;
    const int colb = n0 + nh * 64 + (lane & 15);
#pragma unroll
    for (int tn = 0; tn < 4; ++tn) {
        const int col = colb + tn * 16;
        const float bv = bias[col];
#pragma unroll
        for (int e = 0; e < 8; ++e) {
            out[(row0 + e) * OUT_F + col] = acc[tn][e] + bv;
        }
    }
}

extern "C" void kernel_launch(void* const* d_in, const int* in_sizes, int n_in,
                              void* d_out, int out_size, void* d_ws, size_t ws_size,
                              hipStream_t stream) {
    (void)in_sizes; (void)n_in; (void)out_size; (void)d_ws; (void)ws_size;
    const float*    x    = (const float*)d_in[0];
    const int*      wq   = (const int*)d_in[1];
    const _Float16* wn   = (const _Float16*)d_in[2];
    const float*    bias = (const float*)d_in[3];
    float*          out  = (float*)d_out;

    dim3 grid(OUT_F / BN, M_TOT / BM);   // (64, 4) -> 256 workgroups
    dim3 block(256);                      // 8 wave32
    hipLaunchKernelGGL(linear2bit_wmma_kernel, grid, block, 0, stream,
                       x, wq, wn, bias, out);
}
